// VectorQuantizer_55774445306146
// MI455X (gfx1250) — matrix operand order repaired and checked by reading, not compiled, not run on hardware
//
#include <hip/hip_runtime.h>
#include <hip/hip_bf16.h>
#include <math.h>

// ---------------- problem constants ----------------
#define N_IN   65536
#define DIM    128
#define K_CB   1024
#define COMMIT 0.25f

// ---------------- vector types ----------------
typedef __attribute__((ext_vector_type(16))) __bf16 v16bf;
typedef __attribute__((ext_vector_type(8)))  float  v8f;
typedef __attribute__((ext_vector_type(4)))  unsigned int   u32x4;
typedef __attribute__((ext_vector_type(4)))  float          f32x4;
typedef __attribute__((ext_vector_type(4)))  unsigned short u16x4;
typedef __attribute__((ext_vector_type(2)))  float          f32x2;

union FragBF {
    u32x4 q[2];   // two 16B chunks
    v16bf f;      // 16 bf16 = one WMMA A/B operand (8 VGPRs)
};

__device__ __forceinline__ unsigned short f2bf_rne(float x) {
    unsigned u = __float_as_uint(x);
    unsigned r = u + 0x7FFFu + ((u >> 16) & 1u);   // round-to-nearest-even
    return (unsigned short)(r >> 16);
}

// ---------------- kernel 1: zero accumulators ----------------
__global__ void vq_zero_kernel(unsigned int* hist, float* acc) {
    int i = blockIdx.x * blockDim.x + threadIdx.x;
    if (i < K_CB) hist[i] = 0u;
    if (i == 0)   acc[0] = 0.0f;
}

// ---------------- kernel 2: inputs f32 -> bf16 ----------------
__global__ void vq_cvt_inputs_kernel(const float* __restrict__ in,
                                     unsigned short* __restrict__ out) {
    int t = blockIdx.x * blockDim.x + threadIdx.x;   // one f32x4 per thread
    const f32x4 v = *(const f32x4*)(in + (size_t)t * 4);
    u16x4 o;
    o.x = f2bf_rne(v.x); o.y = f2bf_rne(v.y);
    o.z = f2bf_rne(v.z); o.w = f2bf_rne(v.w);
    *(u16x4*)(out + (size_t)t * 4) = o;
}

// ---------------- kernel 3: codebook -> bf16 + norms ----------------
__global__ void vq_cvt_codebook_kernel(const float* __restrict__ cb,
                                       unsigned short* __restrict__ cb_bf,
                                       float* __restrict__ cbn) {
    int wave = (blockIdx.x * blockDim.x + threadIdx.x) >> 5;   // one wave per code
    int lane = threadIdx.x & 31;
    if (wave >= K_CB) return;
    const float* row = cb + (size_t)wave * DIM + lane * 4;
    f32x4 v = *(const f32x4*)row;
    u16x4 o;
    o.x = f2bf_rne(v.x); o.y = f2bf_rne(v.y);
    o.z = f2bf_rne(v.z); o.w = f2bf_rne(v.w);
    *(u16x4*)(cb_bf + (size_t)wave * DIM + lane * 4) = o;
    float s = v.x*v.x + v.y*v.y + v.z*v.z + v.w*v.w;
    for (int off = 16; off; off >>= 1) s += __shfl_xor(s, off, 32);
    if (lane == 0) cbn[wave] = s;
}

// ---------------- kernel 4: WMMA distance + argmin ----------------
// One wave handles a 16-row tile of inputs vs all 1024 codes.
// score(n) = ||c_n||^2 - 2 * <x, c_n>   (||x||^2 dropped: constant per row)
__global__ void vq_argmin_wmma_kernel(const unsigned short* __restrict__ in_bf,
                                      const unsigned short* __restrict__ cb_bf,
                                      const float* __restrict__ cbn,
                                      int* __restrict__ idx_out) {
    const int wave = (blockIdx.x * blockDim.x + threadIdx.x) >> 5;
    const int lane = threadIdx.x & 31;
    const int m0   = wave * 16;                 // first input row of this tile
    const int lg   = lane & 15;                 // position within half-wave
    const int h    = lane >> 4;                 // which half-wave

    // --- load A fragments (16x32 bf16 per k-step), resident across n-loop ---
    // 16-bit A layout: lanes 0-15 -> M=lane, K in {0..7} U {16..23} (+k0);
    //                  lanes 16-31 -> M=lane-16, K in {8..15} U {24..31}.
    FragBF afr[4];
    const unsigned short* arow = in_bf + (size_t)(m0 + lg) * DIM;
#pragma unroll
    for (int kk = 0; kk < 4; ++kk) {
        const int k0 = kk * 32;
        afr[kk].q[0] = *(const u32x4*)(arow + k0 + 8 * h);
        afr[kk].q[1] = *(const u32x4*)(arow + k0 + 16 + 8 * h);
    }

    float best[8];
    int   bidx[8];
#pragma unroll
    for (int v = 0; v < 8; ++v) { best[v] = 3.4e38f; bidx[v] = 0; }

    // --- loop over code tiles of 16 ---
    for (int n0 = 0; n0 < K_CB; n0 += 16) {
        v8f c = {};
        const unsigned short* brow = cb_bf + (size_t)(n0 + lg) * DIM + 8 * h;
        // B layout (32x16, K-major): lane l<16 -> N=l, K in {0..7} U {16..23} (+k0);
        //                            lane l>=16 -> N=l-16, K in {8..15} U {24..31}.
#pragma unroll
        for (int kk = 0; kk < 4; ++kk) {
            FragBF b;
            b.q[0] = *(const u32x4*)(brow + kk * 32);
            b.q[1] = *(const u32x4*)(brow + kk * 32 + 16);
            c = __builtin_amdgcn_wmma_f32_16x16x32_bf16(
                    false, afr[kk].f, false, b.f, (short)0, c, false, false);
        }
        const float norm = cbn[n0 + lg];        // ||c_N||^2, N = n0 + (lane&15)
        const int   nidx = n0 + lg;
#pragma unroll
        for (int v = 0; v < 8; ++v) {
            float s = norm - 2.0f * c[v];       // C/D layout: M=v+8*h, N=lane&15
            if (s < best[v]) { best[v] = s; bidx[v] = nidx; }
        }
    }

    // --- reduce argmin across the 16 lanes that share each row M ---
#pragma unroll
    for (int off = 8; off; off >>= 1) {
#pragma unroll
        for (int v = 0; v < 8; ++v) {
            float ov = __shfl_xor(best[v], off, 16);
            int   oi = __shfl_xor(bidx[v], off, 16);
            if (ov < best[v] || (ov == best[v] && oi < bidx[v])) {
                best[v] = ov; bidx[v] = oi;
            }
        }
    }
    if (lg == 0) {
#pragma unroll
        for (int v = 0; v < 8; ++v)
            idx_out[m0 + v + 8 * h] = bidx[v];  // lane0 -> M=v, lane16 -> M=v+8
    }
}

// ---------------- kernel 5: one-hot + quantized + stats ----------------
// One wave per input row.
__global__ void vq_scatter_kernel(const float* __restrict__ in,
                                  const float* __restrict__ cb,
                                  const int* __restrict__ idx,
                                  float* __restrict__ out,     // d_out base
                                  unsigned int* __restrict__ hist,
                                  float* __restrict__ acc) {
    const int row  = (blockIdx.x * blockDim.x + threadIdx.x) >> 5;
    const int lane = threadIdx.x & 31;
    const int id   = idx[row];

    // encodings: out[2 + N*D + row*K .. ], base is 8-byte aligned -> f32x2 stores
    float* enc = out + 2 + (size_t)N_IN * DIM + (size_t)row * K_CB;
#pragma unroll
    for (int i = 0; i < 16; ++i) {
        int e  = (lane + 32 * i) * 2;
        f32x2 v;
        v.x = (e     == id) ? 1.0f : 0.0f;
        v.y = (e + 1 == id) ? 1.0f : 0.0f;
        *(f32x2*)(enc + e) = v;
    }

    // quantized_st == codebook[id] numerically; out+1 is only 4B aligned -> scalar
    const float* crow = cb + (size_t)id * DIM;
    const float* xrow = in + (size_t)row * DIM;
    float* qrow = out + 1 + (size_t)row * DIM;
    float ssd = 0.0f;
#pragma unroll
    for (int j = 0; j < 4; ++j) {
        int col = lane + 32 * j;
        float q = crow[col];
        float x = xrow[col];
        qrow[col] = q;
        float d = q - x;
        ssd += d * d;
    }
    for (int off = 16; off; off >>= 1) ssd += __shfl_xor(ssd, off, 32);
    if (lane == 0) {
        atomicAdd(acc, ssd);
        atomicAdd(hist + id, 1u);
    }
}

// ---------------- kernel 6: scalars (loss, perplexity) ----------------
__global__ void vq_finalize_kernel(const unsigned int* __restrict__ hist,
                                   const float* __restrict__ acc,
                                   float* __restrict__ out) {
    __shared__ float sh[1024];
    int t = threadIdx.x;
    float p = (float)hist[t] * (1.0f / (float)N_IN);
    sh[t] = p * logf(p + 1e-10f);
    __syncthreads();
    for (int s = 512; s; s >>= 1) {
        if (t < s) sh[t] += sh[t + s];
        __syncthreads();
    }
    if (t == 0) {
        // q_latent_loss == e_latent_loss numerically -> (1 + COMMIT) * MSE
        out[0] = (1.0f + COMMIT) * acc[0] / ((float)N_IN * (float)DIM);
        out[1 + (size_t)N_IN * DIM] = expf(-sh[0]);   // perplexity
    }
}

// ---------------- launch ----------------
extern "C" void kernel_launch(void* const* d_in, const int* in_sizes, int n_in,
                              void* d_out, int out_size, void* d_ws, size_t ws_size,
                              hipStream_t stream) {
    const float* in  = (const float*)d_in[0];   // [65536,128] f32
    const float* cb  = (const float*)d_in[1];   // [1024,128]  f32
    float* out = (float*)d_out;

    // workspace layout (all offsets 4KB-aligned)
    char* ws = (char*)d_ws;
    unsigned short* in_bf = (unsigned short*)ws;                 // 16 MB
    size_t off = (size_t)N_IN * DIM * sizeof(unsigned short);
    off = (off + 4095) & ~(size_t)4095;
    unsigned short* cb_bf = (unsigned short*)(ws + off);         // 256 KB
    off += (size_t)K_CB * DIM * sizeof(unsigned short);
    off = (off + 4095) & ~(size_t)4095;
    float* cbn = (float*)(ws + off);                             // 4 KB
    off += K_CB * sizeof(float); off = (off + 4095) & ~(size_t)4095;
    int* idx = (int*)(ws + off);                                 // 256 KB
    off += (size_t)N_IN * sizeof(int); off = (off + 4095) & ~(size_t)4095;
    unsigned int* hist = (unsigned int*)(ws + off);              // 4 KB
    off += K_CB * sizeof(unsigned int); off = (off + 4095) & ~(size_t)4095;
    float* acc = (float*)(ws + off);                             // 4 B

    vq_zero_kernel<<<4, 256, 0, stream>>>(hist, acc);
    vq_cvt_inputs_kernel<<<(N_IN * DIM / 4) / 256, 256, 0, stream>>>(in, in_bf);
    vq_cvt_codebook_kernel<<<(K_CB * 32) / 128, 128, 0, stream>>>(cb, cb_bf, cbn);
    // 4096 M-tiles, 4 waves (128 threads) per block
    vq_argmin_wmma_kernel<<<(N_IN / 16) / 4, 128, 0, stream>>>(in_bf, cb_bf, cbn, idx);
    // one wave per row, 8 waves per block
    vq_scatter_kernel<<<N_IN / 8, 256, 0, stream>>>(in, cb, idx, out, hist, acc);
    vq_finalize_kernel<<<1, 1024, 0, stream>>>(hist, acc, out);
}
